// SocialGrouping_model_1606317769311
// MI455X (gfx1250) — matrix operand order repaired and checked by reading, not compiled, not run on hardware
//
#include <hip/hip_runtime.h>
#include <hip/hip_bf16.h>

typedef __attribute__((ext_vector_type(16))) _Float16 v16h;
typedef __attribute__((ext_vector_type(8)))  float    v8f;

// ---------------------------------------------------------------------------
// WMMA fragment loaders (layouts per CDNA5 ISA 7.12.2), f32 -> f16 on the fly.
// fast: unit K-stride, 16B-aligned rows, K-step fully in range -> 4x b128 loads
// slow: arbitrary strides, clamped addresses + post-load zeroing (branch-free)
// ---------------------------------------------------------------------------
__device__ __forceinline__ v16h loadA_fast(const float* __restrict__ A, int ars,
                                           int m0, int k0, int M) {
  const int lane = threadIdx.x & 31;
  int m = m0 + (lane & 15);
  m = (m < M) ? m : (M - 1); // clamp: garbage rows only feed masked-out stores
  const int h8 = (lane >> 4) << 3;
  const float* p = A + (long long)m * ars + k0 + h8;
  const float4 x0 = *(const float4*)(p);
  const float4 x1 = *(const float4*)(p + 4);
  const float4 y0 = *(const float4*)(p + 16);
  const float4 y1 = *(const float4*)(p + 20);
  v16h r;
  r[0] = (_Float16)x0.x;  r[1] = (_Float16)x0.y;  r[2] = (_Float16)x0.z;  r[3] = (_Float16)x0.w;
  r[4] = (_Float16)x1.x;  r[5] = (_Float16)x1.y;  r[6] = (_Float16)x1.z;  r[7] = (_Float16)x1.w;
  r[8] = (_Float16)y0.x;  r[9] = (_Float16)y0.y;  r[10] = (_Float16)y0.z; r[11] = (_Float16)y0.w;
  r[12] = (_Float16)y1.x; r[13] = (_Float16)y1.y; r[14] = (_Float16)y1.z; r[15] = (_Float16)y1.w;
  return r;
}

__device__ __forceinline__ v16h loadB_fast(const float* __restrict__ B, int brs,
                                           int n0, int k0, int N) {
  const int lane = threadIdx.x & 31;
  int n = n0 + (lane & 15);
  n = (n < N) ? n : (N - 1);
  const float* p = B + (long long)n * brs + k0 + ((lane >> 4) << 3);
  const float4 x0 = *(const float4*)(p);
  const float4 x1 = *(const float4*)(p + 4);
  const float4 x2 = *(const float4*)(p + 16);
  const float4 x3 = *(const float4*)(p + 20);
  v16h r;
  r[0] = (_Float16)x0.x;  r[1] = (_Float16)x0.y;  r[2] = (_Float16)x0.z;  r[3] = (_Float16)x0.w;
  r[4] = (_Float16)x1.x;  r[5] = (_Float16)x1.y;  r[6] = (_Float16)x1.z;  r[7] = (_Float16)x1.w;
  r[8] = (_Float16)x2.x;  r[9] = (_Float16)x2.y;  r[10] = (_Float16)x2.z; r[11] = (_Float16)x2.w;
  r[12] = (_Float16)x3.x; r[13] = (_Float16)x3.y; r[14] = (_Float16)x3.z; r[15] = (_Float16)x3.w;
  return r;
}

__device__ __forceinline__ v16h loadA_slow(const float* __restrict__ A, int ars, int acs,
                                           int m0, int k0, int M, int K) {
  const int lane = threadIdx.x & 31;
  const int m = m0 + (lane & 15);
  const int h8 = (lane >> 4) << 3;
  const bool mok = (m < M);
  const long long rowbase = (long long)(mok ? m : 0) * ars;
  v16h r;
#pragma unroll
  for (int e = 0; e < 16; ++e) {
    const int k = k0 + ((e & 8) << 1) + h8 + (e & 7);
    const bool ok = mok && (k < K);
    const float v = A[rowbase + (long long)(ok ? k : 0) * acs];
    r[e] = ok ? (_Float16)v : (_Float16)0.0f;
  }
  return r;
}

__device__ __forceinline__ v16h loadB_slow(const float* __restrict__ B, int brs, int bcs,
                                           int n0, int k0, int N, int K) {
  const int lane = threadIdx.x & 31;
  const int n = n0 + (lane & 15);
  const int kh = k0 + ((lane >> 4) << 3);
  const bool nok = (n < N);
  const long long rowbase = (long long)(nok ? n : 0) * brs;
  v16h r;
#pragma unroll
  for (int e = 0; e < 16; ++e) {
    const int k = kh + ((e & 8) << 1) + (e & 7);
    const bool ok = nok && (k < K);
    const float v = B[rowbase + (long long)(ok ? k : 0) * bcs];
    r[e] = ok ? (_Float16)v : (_Float16)0.0f;
  }
  return r;
}

__device__ __forceinline__ void store_tile(float* __restrict__ Y, int yrs, int ycs,
                                           const float* __restrict__ bias,
                                           const float* __restrict__ res, int rrs, int rcs,
                                           int tm0, int tn0, int M, int N,
                                           v8f acc, float alpha, int act) {
  const int lane = threadIdx.x & 31;
  const int n = tn0 + (lane & 15);
  if (n >= N) return;
  const float bv = bias ? bias[n] : 0.0f;
  const int mb = tm0 + ((lane >> 4) << 3);
#pragma unroll
  for (int r = 0; r < 8; ++r) {
    const int m = mb + r;
    if (m < M) {
      float v = acc[r] * alpha + bv;
      if (res) v += res[(long long)m * rrs + (long long)n * rcs];
      if (act) v = fmaxf(v, 0.0f);
      Y[(long long)m * yrs + (long long)n * ycs] = v;
    }
  }
}

// ---------------------------------------------------------------------------
// Generic batched strided GEMM:  Y[b][m,n] = act(alpha * sum_k A[m,k]*B[n,k]
//                                            + bias[n] + res[m,n])
// One wave per 32x32 output tile: 2x2 register tiling, 4 independent WMMA
// chains per K-step; FA/FB select the vectorized b128 fragment path.
// ---------------------------------------------------------------------------
template <bool FA, bool FB>
__global__ void wmma_gemm(const float* __restrict__ A, long long abatch, int ars, int acs,
                          const float* __restrict__ B, long long bbatch, int brs, int bcs,
                          const float* __restrict__ bias,
                          const float* __restrict__ res, int rrs, int rcs,
                          float* __restrict__ Y, long long ybatch, int yrs, int ycs,
                          int M, int N, int K, float alpha, int act) {
  const int b = blockIdx.z;
  A += (long long)b * abatch;
  B += (long long)b * bbatch;
  Y += (long long)b * ybatch;
  const int m0 = blockIdx.y * 32;
  const int n0 = blockIdx.x * 32;

  v8f acc00 = {}, acc01 = {}, acc10 = {}, acc11 = {};
  const int kmain = K & ~31;
  for (int k0 = 0; k0 < kmain; k0 += 32) {
    const v16h a0 = FA ? loadA_fast(A, ars, m0, k0, M)
                       : loadA_slow(A, ars, acs, m0, k0, M, K);
    const v16h a1 = FA ? loadA_fast(A, ars, m0 + 16, k0, M)
                       : loadA_slow(A, ars, acs, m0 + 16, k0, M, K);
    const v16h b0 = FB ? loadB_fast(B, brs, n0, k0, N)
                       : loadB_slow(B, brs, bcs, n0, k0, N, K);
    const v16h b1 = FB ? loadB_fast(B, brs, n0 + 16, k0, N)
                       : loadB_slow(B, brs, bcs, n0 + 16, k0, N, K);
    acc00 = __builtin_amdgcn_wmma_f32_16x16x32_f16(false, a0, false, b0, (short)0, acc00, false, false);
    acc01 = __builtin_amdgcn_wmma_f32_16x16x32_f16(false, a0, false, b1, (short)0, acc01, false, false);
    acc10 = __builtin_amdgcn_wmma_f32_16x16x32_f16(false, a1, false, b0, (short)0, acc10, false, false);
    acc11 = __builtin_amdgcn_wmma_f32_16x16x32_f16(false, a1, false, b1, (short)0, acc11, false, false);
  }
  if (kmain < K) { // masked K tail
    const v16h a0 = loadA_slow(A, ars, acs, m0, kmain, M, K);
    const v16h a1 = loadA_slow(A, ars, acs, m0 + 16, kmain, M, K);
    const v16h b0 = loadB_slow(B, brs, bcs, n0, kmain, N, K);
    const v16h b1 = loadB_slow(B, brs, bcs, n0 + 16, kmain, N, K);
    acc00 = __builtin_amdgcn_wmma_f32_16x16x32_f16(false, a0, false, b0, (short)0, acc00, false, false);
    acc01 = __builtin_amdgcn_wmma_f32_16x16x32_f16(false, a0, false, b1, (short)0, acc01, false, false);
    acc10 = __builtin_amdgcn_wmma_f32_16x16x32_f16(false, a1, false, b0, (short)0, acc10, false, false);
    acc11 = __builtin_amdgcn_wmma_f32_16x16x32_f16(false, a1, false, b1, (short)0, acc11, false, false);
  }

  const bool do_m1 = (m0 + 16) < M;
  const bool do_n1 = (n0 + 16) < N;
  store_tile(Y, yrs, ycs, bias, res, rrs, rcs, m0, n0, M, N, acc00, alpha, act);
  if (do_n1) store_tile(Y, yrs, ycs, bias, res, rrs, rcs, m0, n0 + 16, M, N, acc01, alpha, act);
  if (do_m1) store_tile(Y, yrs, ycs, bias, res, rrs, rcs, m0 + 16, n0, M, N, acc10, alpha, act);
  if (do_m1 && do_n1) store_tile(Y, yrs, ycs, bias, res, rrs, rcs, m0 + 16, n0 + 16, M, N, acc11, alpha, act);
}

// ---------------------------------------------------------------------------
// feat[i,c] = (1/25) * sum_p ( z_t[i][p][c] + xf[i][c][p] )
// ---------------------------------------------------------------------------
__global__ void feat_kernel(const float* __restrict__ zt, const float* __restrict__ xf,
                            float* __restrict__ feat) {
  const int idx = blockIdx.x * blockDim.x + threadIdx.x;
  if (idx >= 320 * 832) return;
  const int i = idx / 832, c = idx % 832;
  const float* z = zt + (long long)i * 20800 + c;                 // stride 832 over p
  const float* x = xf + (long long)i * 20800 + (long long)c * 25; // stride 1 over p
  float s = 0.0f;
#pragma unroll
  for (int p = 0; p < 25; ++p) s += z[p * 832] + x[p];
  feat[idx] = s * 0.04f;
}

// ---------------------------------------------------------------------------
// GAT attention logits: e[i,j,h] = sum_c lrelu(xr[i,h,c]+xl[j,h,c],0.2)*att[h,c]
// block = (i,h); 8 waves sweep j; xr row + att row staged in LDS
// ---------------------------------------------------------------------------
__global__ void gat_e_kernel(const float* __restrict__ xl, const float* __restrict__ xr,
                             const float* __restrict__ att, float* __restrict__ e) {
  const int i = blockIdx.x, h = blockIdx.y;
  __shared__ float sxr[512];
  __shared__ float satt[512];
  for (int c = threadIdx.x; c < 512; c += blockDim.x) {
    sxr[c]  = xr[(long long)i * 4096 + h * 512 + c];
    satt[c] = att[h * 512 + c];
  }
  __syncthreads();
  const int wave = threadIdx.x >> 5, lane = threadIdx.x & 31;
  for (int j = wave; j < 320; j += (blockDim.x >> 5)) {
    const float* xlj = xl + (long long)j * 4096 + h * 512;
    float s = 0.0f;
    for (int c = lane; c < 512; c += 32) {
      const float t = sxr[c] + xlj[c];
      s += (fmaxf(t, 0.0f) + 0.2f * fminf(t, 0.0f)) * satt[c];
    }
#pragma unroll
    for (int o = 16; o; o >>= 1) s += __shfl_xor(s, o, 32);
    if (lane == 0) e[((long long)i * 320 + j) * 8 + h] = s;
  }
}

// softmax over `len` elements at base = bx*s1 + by*s2, element stride es (in place)
__global__ void softmax_rows(float* __restrict__ d, long long s1, long long s2,
                             long long es, int len) {
  const long long base = (long long)blockIdx.x * s1 + (long long)blockIdx.y * s2;
  const int lane = threadIdx.x;
  float mx = -3.0e38f;
  for (int i = lane; i < len; i += 32) mx = fmaxf(mx, d[base + i * es]);
#pragma unroll
  for (int o = 16; o; o >>= 1) mx = fmaxf(mx, __shfl_xor(mx, o, 32));
  float sum = 0.0f;
  for (int i = lane; i < len; i += 32) {
    const float v = __expf(d[base + i * es] - mx);
    d[base + i * es] = v;
    sum += v;
  }
#pragma unroll
  for (int o = 16; o; o >>= 1) sum += __shfl_xor(sum, o, 32);
  const float inv = 1.0f / sum;
  for (int i = lane; i < len; i += 32) d[base + i * es] *= inv;
}

// gat[i,c] = (1/8) sum_h gatH[h][i,c] + bias[c]
__global__ void gat_mean(const float* __restrict__ gatH, const float* __restrict__ bias,
                         float* __restrict__ gat) {
  const int idx = blockIdx.x * blockDim.x + threadIdx.x;
  if (idx >= 320 * 512) return;
  float s = 0.0f;
#pragma unroll
  for (int h = 0; h < 8; ++h) s += gatH[(long long)h * 163840 + idx];
  gat[idx] = s * 0.125f + bias[idx % 512];
}

// src = concat(group_token, gat)
__global__ void build_src(const float* __restrict__ gtok, const float* __restrict__ gat,
                          float* __restrict__ x) {
  const int idx = blockIdx.x * blockDim.x + threadIdx.x;
  if (idx >= 322 * 512) return;
  x[idx] = (idx < 1024) ? gtok[idx] : gat[idx - 1024];
}

// layernorm over D elements, one wave per row
__global__ void layernorm_k(const float* __restrict__ in, const float* __restrict__ w,
                            const float* __restrict__ b, float* __restrict__ out, int D) {
  const long long base = (long long)blockIdx.x * D;
  const int lane = threadIdx.x;
  float s = 0.0f;
  for (int i = lane; i < D; i += 32) s += in[base + i];
#pragma unroll
  for (int o = 16; o; o >>= 1) s += __shfl_xor(s, o, 32);
  const float mu = s / (float)D;
  float v = 0.0f;
  for (int i = lane; i < D; i += 32) { const float t = in[base + i] - mu; v += t * t; }
#pragma unroll
  for (int o = 16; o; o >>= 1) v += __shfl_xor(v, o, 32);
  const float inv = rsqrtf(v / (float)D + 1e-5f);
  for (int i = lane; i < D; i += 32)
    out[base + i] = (in[base + i] - mu) * inv * w[i] + b[i];
}

// inp[i] = concat(tokens(1024), features_i(512), bboxes_i/IMG_WH(4))
__global__ void build_inp(const float* __restrict__ xenc, const float* __restrict__ bboxes,
                          float* __restrict__ inp) {
  const int idx = blockIdx.x * blockDim.x + threadIdx.x;
  if (idx >= 320 * 1540) return;
  const int i = idx / 1540, c = idx % 1540;
  float v;
  if (c < 1024) v = xenc[c];
  else if (c < 1536) v = xenc[(long long)(2 + i) * 512 + (c - 1024)];
  else {
    const float wh[4] = {1280.0f, 720.0f, 1280.0f, 720.0f};
    v = bboxes[i * 4 + (c - 1536)] / wh[c - 1536];
  }
  inp[(long long)i * 1540 + c] = v;
}

// sim from Gram matrix P = phi*phi^T
__global__ void sim_kernel(const float* __restrict__ P, float* __restrict__ out) {
  const int idx = blockIdx.x * blockDim.x + threadIdx.x;
  if (idx >= 320 * 320) return;
  const int i = idx / 320, j = idx % 320;
  if (i == j) { out[idx] = 1.0f; return; }
  const float d2 = fmaxf(P[i * 321] + P[j * 321] - 2.0f * P[idx], 0.0f);
  out[idx] = __expf(-d2 * (1.0f / 200.0f)); // 2*SIGMA^2 = 200
}

// ---------------------------------------------------------------------------
static void gemm(hipStream_t st,
                 const float* A, long long ab, int ars, int acs,
                 const float* B, long long bb, int brs, int bcs,
                 const float* bias, const float* res, int rrs, int rcs,
                 float* Y, long long yb, int yrs, int ycs,
                 int M, int N, int K, int batch, float alpha, int act) {
  dim3 g((N + 31) / 32, (M + 31) / 32, batch);
  const bool fa = (acs == 1) && (ars % 4 == 0);
  const bool fb = (bcs == 1) && (brs % 4 == 0);
  if (fa && fb)
    wmma_gemm<true, true><<<g, 32, 0, st>>>(A, ab, ars, acs, B, bb, brs, bcs, bias,
                                            res, rrs, rcs, Y, yb, yrs, ycs, M, N, K, alpha, act);
  else if (fa)
    wmma_gemm<true, false><<<g, 32, 0, st>>>(A, ab, ars, acs, B, bb, brs, bcs, bias,
                                             res, rrs, rcs, Y, yb, yrs, ycs, M, N, K, alpha, act);
  else if (fb)
    wmma_gemm<false, true><<<g, 32, 0, st>>>(A, ab, ars, acs, B, bb, brs, bcs, bias,
                                             res, rrs, rcs, Y, yb, yrs, ycs, M, N, K, alpha, act);
  else
    wmma_gemm<false, false><<<g, 32, 0, st>>>(A, ab, ars, acs, B, bb, brs, bcs, bias,
                                              res, rrs, rcs, Y, yb, yrs, ycs, M, N, K, alpha, act);
}

extern "C" void kernel_launch(void* const* d_in, const int* in_sizes, int n_in,
                              void* d_out, int out_size, void* d_ws, size_t ws_size,
                              hipStream_t stream) {
  (void)in_sizes; (void)n_in; (void)out_size; (void)ws_size;
  const float* boxes    = (const float*)d_in[0];
  const float* bboxes   = (const float*)d_in[1];
  const float* W_theta  = (const float*)d_in[2];
  const float* b_theta  = (const float*)d_in[3];
  const float* W_phi    = (const float*)d_in[4];
  const float* b_phi    = (const float*)d_in[5];
  const float* W_g      = (const float*)d_in[6];
  const float* b_g      = (const float*)d_in[7];
  const float* W_z      = (const float*)d_in[8];
  const float* b_z      = (const float*)d_in[9];
  const float* W_emb    = (const float*)d_in[10];
  const float* b_emb    = (const float*)d_in[11];
  const float* gat_Wl   = (const float*)d_in[12];
  const float* gat_bl   = (const float*)d_in[13];
  const float* gat_Wr   = (const float*)d_in[14];
  const float* gat_br   = (const float*)d_in[15];
  const float* gat_att  = (const float*)d_in[16];
  const float* gat_bias = (const float*)d_in[17];
  const float* gtok     = (const float*)d_in[18];
  const float* eWq  = (const float*)d_in[19];
  const float* ebq  = (const float*)d_in[20];
  const float* eWk  = (const float*)d_in[21];
  const float* ebk  = (const float*)d_in[22];
  const float* eWv  = (const float*)d_in[23];
  const float* ebv  = (const float*)d_in[24];
  const float* eWo  = (const float*)d_in[25];
  const float* ebo  = (const float*)d_in[26];
  const float* eln1w = (const float*)d_in[27];
  const float* eln1b = (const float*)d_in[28];
  const float* eW1  = (const float*)d_in[29];
  const float* eb1  = (const float*)d_in[30];
  const float* eW2  = (const float*)d_in[31];
  const float* eb2  = (const float*)d_in[32];
  const float* eln2w = (const float*)d_in[33];
  const float* eln2b = (const float*)d_in[34];
  const float* pW1  = (const float*)d_in[35];
  const float* pb1  = (const float*)d_in[36];
  const float* pW2  = (const float*)d_in[37];
  const float* pb2  = (const float*)d_in[38];
  const float* pW3  = (const float*)d_in[39];
  const float* pb3  = (const float*)d_in[40];

  float* ws = (float*)d_ws;
  size_t off = 0;
  auto alloc = [&](size_t n) { float* p = ws + off; off += n; return p; };

  // persistent across phases
  float* feat = alloc(320 * 832);
  const size_t mark = off;

  // ---- Phase A: per-person non-local block (320 batched small GEMMs) ----
  float* th_t = alloc(832000);   // (i)(25 x 104)
  float* ph_t = alloc(832000);
  float* g_t  = alloc(832000);
  float* fm   = alloc(200000);   // (i)(25 x 25)
  float* y_t  = alloc(832000);
  float* z_t  = alloc(6656000);  // (i)(25 x 832)

  // th_t[i][p,d] = sum_c xf[i][c,p]*W_theta[d,c] + b_theta[d]
  gemm(stream, boxes, 20800, 1, 25,  W_theta, 0, 832, 1, b_theta, nullptr, 0, 0,
       th_t, 2600, 104, 1, 25, 104, 832, 320, 1.0f, 0);
  gemm(stream, boxes, 20800, 1, 25,  W_phi,   0, 832, 1, b_phi,   nullptr, 0, 0,
       ph_t, 2600, 104, 1, 25, 104, 832, 320, 1.0f, 0);
  gemm(stream, boxes, 20800, 1, 25,  W_g,     0, 832, 1, b_g,     nullptr, 0, 0,
       g_t,  2600, 104, 1, 25, 104, 832, 320, 1.0f, 0);
  // f[i][p,q] = (1/25) th_t[p,:]·ph_t[q,:]
  gemm(stream, th_t, 2600, 104, 1,  ph_t, 2600, 104, 1, nullptr, nullptr, 0, 0,
       fm, 625, 25, 1, 25, 25, 104, 320, 1.0f / 25.0f, 0);
  // y_t[i][p,d] = sum_q f[p,q] g_t[q,d]
  gemm(stream, fm, 625, 25, 1,  g_t, 2600, 1, 104, nullptr, nullptr, 0, 0,
       y_t, 2600, 104, 1, 25, 104, 25, 320, 1.0f, 0);
  // z_t[i][p,c] = sum_d y_t[p,d] W_z[c,d] + b_z[c]
  gemm(stream, y_t, 2600, 104, 1,  W_z, 0, 104, 1, b_z, nullptr, 0, 0,
       z_t, 20800, 832, 1, 25, 832, 104, 320, 1.0f, 0);
  feat_kernel<<<(320 * 832 + 255) / 256, 256, 0, stream>>>(z_t, boxes, feat);

  // ---- Phase B: reuse phase-A scratch ----
  off = mark;
  float* emb  = alloc(163840);
  float* xl   = alloc(1310720);
  float* xr   = alloc(1310720);
  float* ee   = alloc(819200);   // (i,j,h)
  float* gatH = alloc(1310720);  // (h)(320 x 512)
  float* gat  = alloc(163840);

  gemm(stream, feat, 0, 832, 1,  W_emb, 0, 832, 1, b_emb, nullptr, 0, 0,
       emb, 0, 512, 1, 320, 512, 832, 1, 1.0f, 0);
  gemm(stream, emb, 0, 512, 1,  gat_Wl, 0, 512, 1, gat_bl, nullptr, 0, 0,
       xl, 0, 4096, 1, 320, 4096, 512, 1, 1.0f, 0);
  gemm(stream, emb, 0, 512, 1,  gat_Wr, 0, 512, 1, gat_br, nullptr, 0, 0,
       xr, 0, 4096, 1, 320, 4096, 512, 1, 1.0f, 0);
  gat_e_kernel<<<dim3(320, 8), 256, 0, stream>>>(xl, xr, gat_att, ee);
  softmax_rows<<<dim3(320, 8), 32, 0, stream>>>(ee, 2560, 1, 8, 320); // over j
  // gatH[h][i,c] = sum_j alpha[i,j,h] * xl[j, h*512+c]
  gemm(stream, ee, 1, 2560, 8,  xl, 512, 1, 4096, nullptr, nullptr, 0, 0,
       gatH, 163840, 512, 1, 320, 512, 320, 8, 1.0f, 0);
  gat_mean<<<(320 * 512 + 255) / 256, 256, 0, stream>>>(gatH, gat_bias, gat);

  // ---- Encoder: 6 layers on x (322 x 512) ----
  float* x      = alloc(164864);
  float* qb     = alloc(164864);
  float* kb     = alloc(164864);
  float* vb     = alloc(164864);
  float* att    = alloc(829472); // (h)(322 x 322)
  float* attout = alloc(164864);
  float* tmp    = alloc(164864);
  float* xn     = alloc(164864);
  float* ff     = alloc(659456);
  float* tmp2   = alloc(164864);

  build_src<<<(322 * 512 + 255) / 256, 256, 0, stream>>>(gtok, gat, x);

  for (int l = 0; l < 6; ++l) {
    const float* Wq = eWq + (long long)l * 262144;  const float* bq = ebq + l * 512;
    const float* Wk = eWk + (long long)l * 262144;  const float* bk = ebk + l * 512;
    const float* Wv = eWv + (long long)l * 262144;  const float* bv = ebv + l * 512;
    const float* Wo = eWo + (long long)l * 262144;  const float* bo = ebo + l * 512;
    const float* W1 = eW1 + (long long)l * 1048576; const float* b1 = eb1 + l * 2048;
    const float* W2 = eW2 + (long long)l * 1048576; const float* b2 = eb2 + l * 512;

    gemm(stream, x, 0, 512, 1, Wq, 0, 512, 1, bq, nullptr, 0, 0,
         qb, 0, 512, 1, 322, 512, 512, 1, 1.0f, 0);
    gemm(stream, x, 0, 512, 1, Wk, 0, 512, 1, bk, nullptr, 0, 0,
         kb, 0, 512, 1, 322, 512, 512, 1, 1.0f, 0);
    gemm(stream, x, 0, 512, 1, Wv, 0, 512, 1, bv, nullptr, 0, 0,
         vb, 0, 512, 1, 322, 512, 512, 1, 1.0f, 0);
    // scores[h] = q_h k_h^T * (1/sqrt(64))
    gemm(stream, qb, 64, 512, 1, kb, 64, 512, 1, nullptr, nullptr, 0, 0,
         att, 103684, 322, 1, 322, 322, 64, 8, 0.125f, 0);
    softmax_rows<<<dim3(322, 8), 32, 0, stream>>>(att, 322, 103684, 1, 322);
    // attout[:, h*64:(h+1)*64] = att_h @ v_h
    gemm(stream, att, 103684, 322, 1, vb, 64, 1, 512, nullptr, nullptr, 0, 0,
         attout, 64, 512, 1, 322, 64, 322, 8, 1.0f, 0);
    // o-proj + residual
    gemm(stream, attout, 0, 512, 1, Wo, 0, 512, 1, bo, x, 512, 1,
         tmp, 0, 512, 1, 322, 512, 512, 1, 1.0f, 0);
    layernorm_k<<<322, 32, 0, stream>>>(tmp, eln1w + l * 512, eln1b + l * 512, xn, 512);
    gemm(stream, xn, 0, 512, 1, W1, 0, 512, 1, b1, nullptr, 0, 0,
         ff, 0, 2048, 1, 322, 2048, 512, 1, 1.0f, 1);
    gemm(stream, ff, 0, 2048, 1, W2, 0, 2048, 1, b2, xn, 512, 1,
         tmp2, 0, 512, 1, 322, 512, 2048, 1, 1.0f, 0);
    layernorm_k<<<322, 32, 0, stream>>>(tmp2, eln2w + l * 512, eln2b + l * 512, x, 512);
  }

  // ---- Head: phi MLP + RBF similarity ----
  float* inp = alloc(492800);
  float* h1  = alloc(163840);
  float* h2  = alloc(163840);
  float* phi = alloc(81920);
  float* P   = alloc(102400);

  build_inp<<<(320 * 1540 + 255) / 256, 256, 0, stream>>>(x, bboxes, inp);
  gemm(stream, inp, 0, 1540, 1, pW1, 0, 1540, 1, pb1, nullptr, 0, 0,
       h1, 0, 512, 1, 320, 512, 1540, 1, 1.0f, 1);
  gemm(stream, h1, 0, 512, 1, pW2, 0, 512, 1, pb2, nullptr, 0, 0,
       h2, 0, 512, 1, 320, 512, 512, 1, 1.0f, 1);
  gemm(stream, h2, 0, 512, 1, pW3, 0, 512, 1, pb3, nullptr, 0, 0,
       phi, 0, 256, 1, 320, 256, 512, 1, 1.0f, 0);
  // Gram matrix P = phi @ phi^T
  gemm(stream, phi, 0, 256, 1, phi, 0, 256, 1, nullptr, nullptr, 0, 0,
       P, 0, 320, 1, 320, 320, 256, 1, 1.0f, 0);
  sim_kernel<<<(320 * 320 + 255) / 256, 256, 0, stream>>>(P, (float*)d_out);
}
